// IstftDecoder_63496796504211
// MI455X (gfx1250) — matrix operand order repaired and checked by reading, not compiled, not run on hardware
//
#include <hip/hip_runtime.h>

typedef __attribute__((ext_vector_type(16))) _Float16 v16h;
typedef __attribute__((ext_vector_type(8)))  _Float16 v8h;
typedef __attribute__((ext_vector_type(8)))  float    v8f;

#define WIN     400
#define NSHIFT  160
#define NFFT    512
#define NBINS   257
#define KDIM    514      // 257 real + 257 imag channels
#define FRAMES  3000
#define NBATCH  16
#define TOUT    (NSHIFT * (FRAMES - 1) + WIN)   // 480240

#define KPAD    544      // 17 * 32
#define MPAD    512      // 32 m-tiles of 16 (rows >= 400 are zero in basis)
#define BLK_M   256
#define BLK_N   128
#define KSTEP   32
#define NKSTEP  (KPAD / KSTEP)   // 17
#define NMTILE  (MPAD / 16)      // 32
#define SSTR    40               // LDS row stride in halves: 80B -> 16B aligned, bank-spread
#define SBUF    (BLK_N * SSTR)   // one LDS plane buffer (halves)

// A-fragment workspace: chunk(kb, mt, plane) = ((kb*NMTILE + mt)*2 + plane),
// each chunk = 32 lanes x 16 halves (1 KB). Total 17*32*2 KB = 1,114,112 B.
#define AWS_CHUNK_H 512          // halves per chunk

__global__ void zero_kernel(float* p, int n) {
    int i = blockIdx.x * blockDim.x + threadIdx.x;
    if (i < n) p[i] = 0.0f;
}

// Build basis directly in WMMA A-operand fragment order, split into f16 hi/lo planes.
// A 16x32 f16 layout: lane = r + lh*16 (r = M row); v16h position p holds
// K = (p<8) ? lh*8+p : 16+lh*8+(p-8).
__global__ void build_basis(_Float16* __restrict__ aws) {
    int idx = blockIdx.x * blockDim.x + threadIdx.x;
    if (idx >= MPAD * KPAD) return;
    int m = idx / KPAD;          // output sample n
    int k = idx % KPAD;          // concat channel
    float val = 0.0f;
    if (m < WIN && k < KDIM) {
        const float twopi = 6.28318530717958647692f;
        float am = twopi * (float)m / 400.0f;
        float w  = 0.42f - 0.5f * cosf(am) + 0.08f * cosf(2.0f * am);
        float d = 0.0f;
        for (int j = m % NSHIFT; j < WIN; j += NSHIFT) {
            float aj = twopi * (float)j / 400.0f;
            float wj = 0.42f - 0.5f * cosf(aj) + 0.08f * cosf(2.0f * aj);
            d += wj * wj;
        }
        float syn = w / d;
        int  kk2 = (k < NBINS) ? k : (k - NBINS);
        int  ph  = (kk2 * m) & (NFFT - 1);          // exact integer phase reduction
        float ang = (float)ph * (twopi / 512.0f);
        float c   = (k < NBINS) ? cosf(ang) : -sinf(ang);
        float coef = ((kk2 == 0) || (kk2 == NBINS - 1)) ? 1.0f : 2.0f;
        val = coef * (1.0f / 512.0f) * c * syn;
    }
    _Float16 hi = (_Float16)val;
    _Float16 lo = (_Float16)(val - (float)hi);

    int mt = m >> 4;             // m-tile
    int r  = m & 15;             // row within tile
    int kb = k >> 5;             // k-step
    int kk = k & 31;             // K within step
    int lh = (kk >> 3) & 1;      // lane half: lanes 0-15 hold K {0..7}U{16..23}
    int p  = (kk & 7) + ((kk >= 16) ? 8 : 0);
    int lane = r + lh * 16;
    size_t off = (size_t)((kb * NMTILE + mt) * 2) * AWS_CHUNK_H + lane * 16 + p;
    aws[off]               = hi;   // plane 0
    aws[off + AWS_CHUNK_H] = lo;   // plane 1
}

__global__ __launch_bounds__(256) void istft_wmma(
    const float*    __restrict__ stft,   // [16][514][3000]
    const _Float16* __restrict__ aws,    // fragment-ordered basis hi/lo
    float*          __restrict__ out)    // [16][480240], pre-zeroed
{
    // double-buffered stft tile, transposed [n][k], split hi/lo f16
    __shared__ _Float16 sH[2 * SBUF];
    __shared__ _Float16 sL[2 * SBUF];

    const int t    = threadIdx.x;
    const int lane = t & 31;
    const int wave = t >> 5;       // 8 waves
    const int wm   = wave & 3;     // 4 waves along M
    const int wn   = wave >> 2;    // 2 waves along N
    const int lm   = lane & 15;
    const int lh   = lane >> 4;

    const int f_base  = blockIdx.x * BLK_N;
    const int mt_base = blockIdx.y * (BLK_M / 16);   // m-tile base
    const int b       = blockIdx.z;

    const float* Ain = stft + (size_t)b * (size_t)KDIM * (size_t)FRAMES;

    // staging coordinates: thread owns k-pair (2*kp, 2*kp+1) x 8 frames.
    // Guard-free loads via clamping:
    //  - f clamped to FRAMES-8: out-of-range threads stage real finite data into
    //    LDS columns mapping to f>=3000, which the epilogue discards (a GEMM
    //    column only influences that same output column).
    //  - k row-pair base clamped to 512: rows k>=514 get copies of real data,
    //    which multiply an exactly-zero basis row (hi and lo) -> exact zero.
    const int kp  = t >> 4;              // 0..15
    const int fg  = t & 15;              // 0..15
    const int f0  = f_base + fg * 8;
    const int fcl = (f0 <= FRAMES - 8) ? f0 : (FRAMES - 8);

    v8f acc[4][4];
#pragma unroll
    for (int i = 0; i < 4; ++i)
#pragma unroll
        for (int j = 0; j < 4; ++j) acc[i][j] = {};

    // ---- register prefetch of one k-step of raw stft (2 rows x 8 frames) ----
    float4 pre[4];
    {
        int kc = 2 * kp;                           // first step: always < 512
        const float* s0 = Ain + (size_t)kc * FRAMES + fcl;
        pre[0] = *(const float4*)(s0);
        pre[1] = *(const float4*)(s0 + 4);
        pre[2] = *(const float4*)(s0 + FRAMES);
        pre[3] = *(const float4*)(s0 + FRAMES + 4);
    }

    for (int kb = 0; kb < NKSTEP; ++kb) {
        const int buf = (kb & 1) * SBUF;

        // ---- split current prefetch and store to LDS (ping-pong buffer) ----
        {
            _Float16* dH = sH + buf;
            _Float16* dL = sL + buf;
#pragma unroll
            for (int j = 0; j < 2; ++j) {
                float a0[4] = {pre[j].x, pre[j].y, pre[j].z, pre[j].w};
                float a1[4] = {pre[2 + j].x, pre[2 + j].y, pre[2 + j].z, pre[2 + j].w};
#pragma unroll
                for (int e = 0; e < 4; ++e) {
                    int n = fg * 8 + j * 4 + e;
                    union { _Float16 f[2]; unsigned u; } ph, pl;
                    _Float16 h0 = (_Float16)a0[e];
                    _Float16 h1 = (_Float16)a1[e];
                    ph.f[0] = h0;
                    ph.f[1] = h1;
                    pl.f[0] = (_Float16)(a0[e] - (float)h0);
                    pl.f[1] = (_Float16)(a1[e] - (float)h1);
                    *(unsigned*)&dH[n * SSTR + 2 * kp] = ph.u;
                    *(unsigned*)&dL[n * SSTR + 2 * kp] = pl.u;
                }
            }
        }

        // ---- issue next k-step's global loads before the barrier; they retire
        //      during the barrier + WMMA block ----
        if (kb + 1 < NKSTEP) {
            int kg = (kb + 1) * KSTEP + 2 * kp;
            int kc = (kg <= KDIM - 2) ? kg : (KDIM - 2);   // clamp pair base to 512
            const float* s0 = Ain + (size_t)kc * FRAMES + fcl;
            pre[0] = *(const float4*)(s0);
            pre[1] = *(const float4*)(s0 + 4);
            pre[2] = *(const float4*)(s0 + FRAMES);
            pre[3] = *(const float4*)(s0 + FRAMES + 4);
        }
        __syncthreads();

        // ---- B-operand fragments: lane (col n, half lh) holds K = lh*8 + 0..7, 16 + lh*8 + 0..7 ----
        v16h bhi[4], blo[4];
#pragma unroll
        for (int ns = 0; ns < 4; ++ns) {
            int row = buf + (wn * 64 + ns * 16 + lm) * SSTR;
            v8h h0 = *(const v8h*)&sH[row + lh * 8];
            v8h h1 = *(const v8h*)&sH[row + 16 + lh * 8];
            v8h l0 = *(const v8h*)&sL[row + lh * 8];
            v8h l1 = *(const v8h*)&sL[row + 16 + lh * 8];
            bhi[ns] = __builtin_shufflevector(h0, h1, 0,1,2,3,4,5,6,7,8,9,10,11,12,13,14,15);
            blo[ns] = __builtin_shufflevector(l0, l1, 0,1,2,3,4,5,6,7,8,9,10,11,12,13,14,15);
        }

        // ---- A-operand fragments straight from fragment-ordered global ws ----
#pragma unroll
        for (int ms = 0; ms < 4; ++ms) {
            int mt = mt_base + wm * 4 + ms;
            const _Float16* pa = aws + (size_t)((kb * NMTILE + mt) * 2) * AWS_CHUNK_H
                                     + lane * 16;
            v8h a0 = *(const v8h*)pa;
            v8h a1 = *(const v8h*)(pa + 8);
            v8h c0 = *(const v8h*)(pa + AWS_CHUNK_H);
            v8h c1 = *(const v8h*)(pa + AWS_CHUNK_H + 8);
            v16h ahi = __builtin_shufflevector(a0, a1, 0,1,2,3,4,5,6,7,8,9,10,11,12,13,14,15);
            v16h alo = __builtin_shufflevector(c0, c1, 0,1,2,3,4,5,6,7,8,9,10,11,12,13,14,15);
#pragma unroll
            for (int ns = 0; ns < 4; ++ns) {
                acc[ms][ns] = __builtin_amdgcn_wmma_f32_16x16x32_f16(
                    false, ahi, false, bhi[ns], (short)0, acc[ms][ns], false, false);
                acc[ms][ns] = __builtin_amdgcn_wmma_f32_16x16x32_f16(
                    false, ahi, false, blo[ns], (short)0, acc[ms][ns], false, false);
                acc[ms][ns] = __builtin_amdgcn_wmma_f32_16x16x32_f16(
                    false, alo, false, bhi[ns], (short)0, acc[ms][ns], false, false);
            }
        }
    }

    // ---- overlap-add epilogue: out[b][f*160 + n] += frames[n][f] ----
    // C layout: VGPR d: lanes 0-15 -> M=d, N=lane; lanes 16-31 -> M=d+8, N=lane-16.
#pragma unroll
    for (int ms = 0; ms < 4; ++ms) {
        int n0 = (mt_base + wm * 4 + ms) * 16 + lh * 8;
#pragma unroll
        for (int ns = 0; ns < 4; ++ns) {
            int f = f_base + wn * 64 + ns * 16 + lm;
            if (f < FRAMES) {
                float* op = out + (size_t)b * TOUT + (size_t)f * NSHIFT;
#pragma unroll
                for (int d = 0; d < 8; ++d) {
                    int n = n0 + d;
                    if (n < WIN) atomicAdd(op + n, acc[ms][ns][d]);
                }
            }
        }
    }
}

extern "C" void kernel_launch(void* const* d_in, const int* in_sizes, int n_in,
                              void* d_out, int out_size, void* d_ws, size_t ws_size,
                              hipStream_t stream) {
    (void)in_sizes; (void)n_in; (void)ws_size;
    const float* stft = (const float*)d_in[0];
    float* out = (float*)d_out;
    _Float16* aws = (_Float16*)d_ws;   // needs 17*32*2 KB = 1,114,112 bytes

    zero_kernel<<<(out_size + 255) / 256, 256, 0, stream>>>(out, out_size);
    build_basis<<<(MPAD * KPAD) / 256, 256, 0, stream>>>(aws);

    dim3 grid((FRAMES + BLK_N - 1) / BLK_N,   // 24
              MPAD / BLK_M,                   // 2
              NBATCH);                        // 16
    istft_wmma<<<grid, 256, 0, stream>>>(stft, aws, out);
}
